// vMF_fast_38852274160109
// MI455X (gfx1250) — matrix operand order repaired and checked by reading, not compiled, not run on hardware
//
#include <hip/hip_runtime.h>
#include <stdint.h>

typedef __attribute__((ext_vector_type(16))) _Float16 v16h;
typedef __attribute__((ext_vector_type(8)))  _Float16 v8h;
typedef __attribute__((ext_vector_type(8)))  float    v8f;
typedef __attribute__((ext_vector_type(4)))  int      v4i;

#define HDIM 512
#define DDIM 128
#define BM   16          // rows per workgroup
#define KCH  128         // K chunk for async double-buffering (4 chunks)
#define KAPPA 100.0f

#if defined(__HIP_DEVICE_COMPILE__) && \
    __has_builtin(__builtin_amdgcn_global_load_async_to_lds_b128) && \
    __has_builtin(__builtin_amdgcn_s_wait_asynccnt)
#define HAVE_ASYNC 1
#else
#define HAVE_ASYNC 0
#endif

// ---------- RNG helpers (counter-based, deterministic) ----------
__device__ __forceinline__ unsigned pcg(unsigned x) {
    unsigned state = x * 747796405u + 2891336453u;
    unsigned word  = ((state >> ((state >> 28u) + 4u)) ^ state) * 277803737u;
    return (word >> 22u) ^ word;
}
__device__ __forceinline__ float u01(unsigned h) {
    return ((float)h + 0.5f) * 2.3283064365386963e-10f;
}
__device__ __forceinline__ float gauss(unsigned a, unsigned b) {
    float u1 = u01(a), u2 = u01(b);
    float r = sqrtf(-2.0f * __logf(u1));
    return r * __cosf(6.2831853071795864f * u2);
}

// Wood rejection sampler for vMF radial component (dim = 127), beta(63.5,63.5)
// approximated by its moment-matched normal (var = 1/512).
__device__ __forceinline__ float wood_w(unsigned row) {
    const float dim = (float)(DDIM - 1);
    const float b = dim / (sqrtf(4.0f * KAPPA * KAPPA + dim * dim) + 2.0f * KAPPA);
    const float x = (1.0f - b) / (1.0f + b);
    const float c = KAPPA * x + dim * __logf(1.0f - x * x);
    float w = 0.999f;
    bool accepted = false;
    #pragma unroll 1
    for (int it = 0; it < 8; ++it) {
        unsigned h1 = pcg(row * 2654435761u + (unsigned)it * 2246822519u + 1u);
        unsigned h2 = pcg(h1 ^ 0x85ebca6bu);
        unsigned h3 = pcg(h2 ^ 0xc2b2ae35u);
        float n = gauss(h1, h2);
        float z = fminf(fmaxf(0.5f + 0.044194173824f * n, 1e-6f), 1.0f - 1e-6f);
        float wn = (1.0f - (1.0f + b) * z) / (1.0f - (1.0f - b) * z);
        float u = u01(h3);
        bool ok = (KAPPA * wn + dim * __logf(fmaxf(1.0f - x * wn, 1e-30f)) - c) >= __logf(u);
        if (ok && !accepted) { w = wn; accepted = true; }
    }
    return w;
}

__device__ __forceinline__ float rsum16(float v) {
    v += __shfl_xor(v, 1);
    v += __shfl_xor(v, 2);
    v += __shfl_xor(v, 4);
    v += __shfl_xor(v, 8);
    return v;
}

__device__ __forceinline__ v16h cat8(v8h a, v8h b) {
    return __builtin_shufflevector(a, b, 0,1,2,3,4,5,6,7,8,9,10,11,12,13,14,15);
}

// ---------- pre-kernel: split W_mu (f32) into f16 hi/lo planes in workspace ----
__global__ __launch_bounds__(256)
void wsplit_kernel(const float* __restrict__ Wm,
                   _Float16* __restrict__ whi, _Float16* __restrict__ wlo) {
    int i = blockIdx.x * 256 + threadIdx.x;   // DDIM*HDIM elements
    float x = Wm[i];
    _Float16 h = (_Float16)x;
    whi[i] = h;
    wlo[i] = (_Float16)(x - (float)h);
}

// ---------- fused GEMM + vMF epilogue ----------
__global__ __launch_bounds__(256)
void vmf_fused_kernel(const float*    __restrict__ lat,   // [B,512] f32
                      const _Float16* __restrict__ whi,   // [128,512] f16 hi
                      const _Float16* __restrict__ wlo,   // [128,512] f16 lo
                      const float*    __restrict__ bmu,   // [128]
                      const float*    __restrict__ kld,   // [1]
                      float* __restrict__ out_mu,
                      float* __restrict__ out_rn,
                      float* __restrict__ out_kld,
                      float* __restrict__ out_s)
{
    __shared__ _Float16 a_hi[BM][HDIM + 8];   // stride 520 halves: 16B-aligned rows,
    __shared__ _Float16 a_lo[BM][HDIM + 8];   // bank-conflict-free b128 reads
#if HAVE_ASYNC
    __shared__ float rawbuf[2][BM * KCH];     // double-buffered async landing zone
#endif
    __shared__ float c_s[BM][DDIM + 4];

    const int tid  = threadIdx.x;
    const int wave = tid >> 5;                // 0..7, one 16-col N-tile each
    const int lane = tid & 31;
    const int hh   = lane >> 4;               // WMMA lane half
    const int lm   = lane & 15;
    const int row0 = blockIdx.x * BM;

#if HAVE_ASYNC
    // issue one K-chunk (16 rows x 128 K of f32 = 8 KB) as 512 async b128 copies
    auto issue_chunk = [&](int c, int buf) {
        #pragma unroll
        for (int j = 0; j < 2; ++j) {
            int f  = tid + j * 256;           // float4 index within chunk
            int r  = f >> 5;                  // 32 float4 per row-chunk
            int c4 = (f & 31) << 2;
            const float* gp = lat + (size_t)(row0 + r) * HDIM + c * KCH + c4;
            float*       lp = &rawbuf[buf][r * KCH + c4];
            __builtin_amdgcn_global_load_async_to_lds_b128((v4i*)gp, (v4i*)lp, 0, 0);
        }
    };
#endif

    // convert one K-chunk of A to f16 hi/lo (once per WG, not per wave per K-step)
    auto convert_chunk = [&](int c) {
        int e0 = tid * 8;                     // 2048 elems/chunk, 8 per thread
        int r  = e0 >> 7;
        int k  = e0 & 127;
#if HAVE_ASYNC
        const float* src = &rawbuf[c & 1][e0];
#else
        const float* src = lat + (size_t)(row0 + r) * HDIM + c * KCH + k;
#endif
        float x[8];
        #pragma unroll
        for (int i = 0; i < 8; ++i) x[i] = src[i];
        #pragma unroll
        for (int i = 0; i < 8; ++i) {
            _Float16 h = (_Float16)x[i];
            a_hi[r][c * KCH + k + i] = h;
            a_lo[r][c * KCH + k + i] = (_Float16)(x[i] - (float)h);
        }
    };

    const int nrow = wave * 16 + lm;                     // W row == output column
    const _Float16* __restrict__ whrow = whi + ((size_t)nrow << 9);
    const _Float16* __restrict__ wlrow = wlo + ((size_t)nrow << 9);

    v8f acc = {};

#if HAVE_ASYNC
    issue_chunk(0, 0);
#endif
    #pragma unroll 1
    for (int c = 0; c < HDIM / KCH; ++c) {
#if HAVE_ASYNC
        __builtin_amdgcn_s_wait_asynccnt(0);             // chunk c landed
#endif
        __syncthreads();                                 // prior readers done
#if HAVE_ASYNC
        if (c < HDIM / KCH - 1) issue_chunk(c + 1, (c + 1) & 1);  // overlap DMA
#endif
        convert_chunk(c);
        __syncthreads();

        #pragma unroll
        for (int k2 = 0; k2 < KCH; k2 += 32) {
            const int kc = c * KCH + k2;
            if (kc + 32 < HDIM) {                        // -> global_prefetch_b8
                __builtin_prefetch(whrow + kc + 32 + hh * 16, 0, 0);
                __builtin_prefetch(wlrow + kc + 32 + hh * 16, 0, 0);
            }
            // A: lane lm holds M=lm, K = kc+hh*8+[0..7] | kc+16+hh*8+[0..7]
            v16h ah = cat8(*(const v8h*)&a_hi[lm][kc + hh * 8],
                           *(const v8h*)&a_hi[lm][kc + 16 + hh * 8]);
            v16h al = cat8(*(const v8h*)&a_lo[lm][kc + hh * 8],
                           *(const v8h*)&a_lo[lm][kc + 16 + hh * 8]);
            // B: lane lm holds N=lm, K = kc+hh*8+[0..7] | kc+16+hh*8+[0..7]
            const v8h* bhp = (const v8h*)(whrow + kc + hh * 8);
            const v8h* blp = (const v8h*)(wlrow + kc + hh * 8);
            v16h bh = cat8(bhp[0], bhp[2]);
            v16h bl = cat8(blp[0], blp[2]);

            // D = A*B + C : hi*hi + hi*lo + lo*hi  (near-fp32 product)
            acc = __builtin_amdgcn_wmma_f32_16x16x32_f16(false, ah, false, bh, (short)0, acc, false, false);
            acc = __builtin_amdgcn_wmma_f32_16x16x32_f16(false, ah, false, bl, (short)0, acc, false, false);
            acc = __builtin_amdgcn_wmma_f32_16x16x32_f16(false, al, false, bh, (short)0, acc, false, false);
        }
    }

    // ---- bias + spill C tile (VGPR r -> M = r + 8*hh, N = lm) ----
    {
        const float bias = bmu[wave * 16 + lm];
        #pragma unroll
        for (int r = 0; r < 8; ++r)
            c_s[r + hh * 8][wave * 16 + lm] = acc[r] + bias;
    }
    __syncthreads();

    // ---- row-wise epilogue: 16 threads per row, 8 columns each ----
    const float kld0 = kld[0];
    const int row  = tid >> 4;
    const int seg  = tid & 15;
    const unsigned grow = (unsigned)(row0 + row);
    const int col0 = seg * 8;

    float m[8];
    float ss = 0.0f;
    #pragma unroll
    for (int i = 0; i < 8; ++i) { m[i] = c_s[row][col0 + i]; ss += m[i] * m[i]; }
    ss = rsum16(ss);
    const float nrm = sqrtf(ss);
    const float inv = 1.0f / nrm;

    #pragma unroll
    for (int i = 0; i < 8; ++i) {
        m[i] *= inv;
        out_mu[(size_t)grow * DDIM + col0 + i] = m[i];
    }
    if (seg == 0) {
        const float d = nrm - 1.0f;
        out_rn[grow]  = d * d;
        out_kld[grow] = kld0;
    }

    float v[8];
    float dotp = 0.0f;
    #pragma unroll
    for (int i = 0; i < 8; ++i) {
        unsigned idx = grow * (unsigned)DDIM + (unsigned)(col0 + i);
        unsigned s0 = pcg(idx * 0x9E3779B9u + 0x1234567u);
        unsigned s1 = pcg(s0 ^ 0xDEADBEEFu);
        v[i] = gauss(s0, s1);
        dotp += m[i] * v[i];
    }
    dotp = rsum16(dotp);

    float oss = 0.0f;
    #pragma unroll
    for (int i = 0; i < 8; ++i) { v[i] -= m[i] * dotp; oss += v[i] * v[i]; }
    oss = rsum16(oss);
    const float oinv = rsqrtf(oss);

    const float w   = wood_w(grow);
    const float fac = sqrtf(fmaxf(0.0f, 1.0f - w * w));
    #pragma unroll
    for (int i = 0; i < 8; ++i)
        out_s[(size_t)grow * DDIM + col0 + i] = v[i] * oinv * fac + m[i] * w;
}

extern "C" void kernel_launch(void* const* d_in, const int* in_sizes, int n_in,
                              void* d_out, int out_size, void* d_ws, size_t ws_size,
                              hipStream_t stream) {
    const float* lat = (const float*)d_in[0];   // [B,512]
    const float* Wm  = (const float*)d_in[1];   // [128,512]
    const float* bmu = (const float*)d_in[2];   // [128]
    const float* kld = (const float*)d_in[3];   // [1]
    // d_in[4] = n_sample == 1 (unused)

    const int B = in_sizes[0] / HDIM;           // 131072

    // workspace: W_mu split into f16 hi/lo planes (2 * 128KB), L2-resident
    _Float16* whi = (_Float16*)d_ws;
    _Float16* wlo = whi + (size_t)DDIM * HDIM;

    float* out     = (float*)d_out;
    float* out_mu  = out;                       // [B,128]
    float* out_rn  = out_mu + (size_t)B * DDIM; // [B]
    float* out_kld = out_rn + B;                // [B]
    float* out_s   = out_kld + B;               // [1,B,128]

    hipLaunchKernelGGL(wsplit_kernel, dim3((DDIM * HDIM) / 256), dim3(256), 0, stream,
                       Wm, whi, wlo);
    hipLaunchKernelGGL(vmf_fused_kernel, dim3(B / BM), dim3(256), 0, stream,
                       lat, whi, wlo, bmu, kld, out_mu, out_rn, out_kld, out_s);
}
